// SeqLSTM_54211077210561
// MI455X (gfx1250) — matrix operand order repaired and checked by reading, not compiled, not run on hardware
//
#include <hip/hip_runtime.h>

typedef __attribute__((ext_vector_type(16))) __bf16 v16bf;
typedef __attribute__((ext_vector_type(8)))  __bf16 v8bf;
typedef __attribute__((ext_vector_type(8)))  float  v8f;

// ---------- helpers ----------

// A-matrix 16x32 bf16 per-lane fragment: lane m+16h holds row m,
// K = [16h, 16h+16) contiguous (matches B fragment order), pairs packed per dword.
__device__ __forceinline__ v16bf load_a16(const __bf16* base) {
  v16bf a;
  *reinterpret_cast<v8bf*>(&a) = *reinterpret_cast<const v8bf*>(base);
  *reinterpret_cast<v8bf*>(reinterpret_cast<char*>(&a) + 16) =
      *reinterpret_cast<const v8bf*>(base + 8);
  return a;
}

// B-matrix 32x16 bf16 per-lane fragment: lane n+16h holds column n,
// K = [16h, 16h+16) contiguous -> one 32-byte contiguous load.
__device__ __forceinline__ v16bf load_b16(const __bf16* base) {
  v16bf b;
  *reinterpret_cast<v8bf*>(&b) = *reinterpret_cast<const v8bf*>(base);
  *reinterpret_cast<v8bf*>(reinterpret_cast<char*>(&b) + 16) =
      *reinterpret_cast<const v8bf*>(base + 8);
  return b;
}

__device__ __forceinline__ v8f wmma_bf16(v16bf a, v16bf b, v8f c) {
  return __builtin_amdgcn_wmma_f32_16x16x32_bf16(false, a, false, b,
                                                 (short)0, c, false, false);
}

// CDNA5 async global->LDS copy (16B). VDST VGPR = LDS byte offset within the
// workgroup allocation; VADDR = 64-bit global address (GV mode). ASYNCcnt.
__device__ __forceinline__ void async_copy_b128(unsigned lds_off, const void* gptr) {
  asm volatile("global_load_async_to_lds_b128 %0, %1, off"
               :: "v"(lds_off), "v"((unsigned long long)gptr)
               : "memory");
}
__device__ __forceinline__ void async_wait0() {
  asm volatile("s_wait_asynccnt 0x0" ::: "memory");
}

// sum-reduce across the 16-lane half of a wave32 (masks < 16 stay in-group)
__device__ __forceinline__ float red16(float v) {
  v += __shfl_xor(v, 1);
  v += __shfl_xor(v, 2);
  v += __shfl_xor(v, 4);
  v += __shfl_xor(v, 8);
  return v;
}

__device__ __forceinline__ float sigm(float x) { return 1.f / (1.f + __expf(-x)); }

// ---------- kernels ----------

__global__ void __launch_bounds__(256)
cvt_bf16_kernel(const float* __restrict__ src, __bf16* __restrict__ dst, long n) {
  long i = (long)blockIdx.x * blockDim.x + threadIdx.x;
  long stride = (long)gridDim.x * blockDim.x;
  for (; i < n; i += stride) dst[i] = (__bf16)src[i];
}

// emb fp32 (output, [b][s][256]) + emb bf16 ([s*64+b][256]) for WMMA A-rows
__global__ void __launch_bounds__(128)
gather_kernel(const int* __restrict__ indices, const float* __restrict__ table,
              float* __restrict__ emb_out, __bf16* __restrict__ emb_sb) {
  int blk = blockIdx.x;            // b*512 + j
  int b = blk >> 9, j = blk & 511;
  int s = j >> 1, hf = j & 1;
  int e = threadIdx.x;             // 0..127
  int idx = indices[b * 512 + j];
  float v = table[(long)idx * 128 + e];
  emb_out[(long)b * 65536 + s * 256 + hf * 128 + e] = v;
  emb_sb[(long)(s * 64 + b) * 256 + hf * 128 + e] = (__bf16)v;
}

// x_gates = LN_g(emb @ W_ih^T): M=16384 (rows = s*64+b), N=1024, K=256.
// One block (8 waves) per 16-row strip. The shared 16x256 A block is staged
// into LDS once via async global->LDS; LN over N fused via LDS.
__global__ void __launch_bounds__(256, 1)
xgates_kernel(const __bf16* __restrict__ emb_sb,
              const __bf16* __restrict__ Wih,
              const float* __restrict__ g_ih,
              const float* __restrict__ b_ih,
              __bf16* __restrict__ xg) {
  extern __shared__ char smem[];
  float*  raw  = (float*)smem;               // 16 x 1024 fp32 (64KB)
  __bf16* abuf = (__bf16*)(smem + 65536);    // 16 x 256 bf16  (8KB)
  const int tid = threadIdx.x;
  const int w = tid >> 5, lane = tid & 31;
  const int m = lane & 15, h = lane >> 4;
  const int r0 = blockIdx.x * 16;
  const int ncol0 = w * 128;

  // async-stage the A block (shared by all 8 waves): 8KB, 32B per thread
  {
    unsigned lo = 65536u + (unsigned)tid * 32u;
    const __bf16* g = emb_sb + (long)r0 * 256 + tid * 16;
    async_copy_b128(lo, g);
    async_copy_b128(lo + 16u, (const char*)g + 16);
  }
  async_wait0();
  __syncthreads();

  v8f acc[8] = {};
#pragma unroll
  for (int ks = 0; ks < 8; ++ks) {
    v16bf a = load_a16(abuf + m * 256 + ks * 32 + 16 * h);
    v16bf bm[8];
#pragma unroll
    for (int t = 0; t < 8; ++t)
      bm[t] = load_b16(Wih + (ncol0 + t * 16 + m) * 256 + ks * 32 + 16 * h);
#pragma unroll
    for (int t = 0; t < 8; ++t)
      acc[t] = wmma_bf16(a, bm[t], acc[t]);
  }
#pragma unroll
  for (int t = 0; t < 8; ++t)
#pragma unroll
    for (int v = 0; v < 8; ++v)
      raw[(v + 8 * h) * 1024 + ncol0 + t * 16 + m] = acc[t][v];
  __syncthreads();

  const int row = tid >> 4, l16 = tid & 15;
  float su = 0.f, ss = 0.f;
  const float* rr = raw + row * 1024 + l16 * 64;
  for (int c = 0; c < 64; ++c) { float v = rr[c]; su += v; ss += v * v; }
  su = red16(su); ss = red16(ss);
  float mean = su * (1.f / 1024.f);
  float rstd = rsqrtf(ss * (1.f / 1024.f) - mean * mean + 1e-5f);
  __bf16* orow = xg + (long)(r0 + row) * 1024;
  for (int c = 0; c < 64; ++c) {
    int col = l16 * 64 + c;
    orow[col] = (__bf16)((raw[row * 1024 + col] - mean) * rstd * g_ih[col] + b_ih[col]);
  }
}

// Persistent recurrence: 1 workgroup, 1024 threads = 32 wave32s on one WGP.
// LDS: gates 64x1024 fp32 (256KB) + h 64x256 bf16 (32KB) = 288KB <= 320KB.
// Each wave owns a 32-column strip of W_hh and computes all 4 M-tiles in two
// groups of 2 (acc pressure 32 VGPRs) -- W_hh is read 2x per step from L2.
__global__ void __launch_bounds__(1024, 1)
lstm_kernel(const __bf16* __restrict__ Whh, const __bf16* __restrict__ xg,
            const float* __restrict__ g_hh, const float* __restrict__ b_hh,
            const float* __restrict__ g_c, const float* __restrict__ b_c,
            const float* __restrict__ h0, const float* __restrict__ c0,
            __bf16* __restrict__ hs) {
  extern __shared__ char smem[];
  float* gates = (float*)smem;                       // 64*1024 fp32
  __bf16* hbuf = (__bf16*)(smem + 64 * 1024 * 4);    // 64*256 bf16
  const int tid = threadIdx.x;
  const int w = tid >> 5, lane = tid & 31;
  const int m = lane & 15, h = lane >> 4;
  const int ncol0 = w * 32;              // 32 waves x 32 cols = 1024
  const int b = tid >> 4, l16 = tid & 15, j0 = l16 * 16;

  for (int i = 0; i < 16; ++i) {
    int idx = tid * 16 + i;
    hbuf[idx] = (__bf16)h0[idx];
  }
  float c[16];
#pragma unroll
  for (int i = 0; i < 16; ++i) c[i] = c0[b * 256 + j0 + i];
  __syncthreads();

  const __bf16* wrow0 = Whh + (ncol0 + m) * 256 + 16 * h;        // u = 0
  const __bf16* wrow1 = Whh + (ncol0 + 16 + m) * 256 + 16 * h;   // u = 1

  for (int s = 0; s < 256; ++s) {
    // ---- GEMM: raw = h @ W_hh^T  (64x1024, K=256, bf16 WMMA) ----
#pragma unroll
    for (int mg = 0; mg < 2; ++mg) {
      v8f acc[4] = {};                   // [mt2][u]
#pragma unroll
      for (int ks = 0; ks < 8; ++ks) {
        v16bf b0 = load_b16(wrow0 + ks * 32);
        v16bf b1 = load_b16(wrow1 + ks * 32);
#pragma unroll
        for (int mt2 = 0; mt2 < 2; ++mt2) {
          int mt = mg * 2 + mt2;
          v16bf a = load_a16(hbuf + (16 * mt + m) * 256 + ks * 32 + 16 * h);
          acc[mt2 * 2 + 0] = wmma_bf16(a, b0, acc[mt2 * 2 + 0]);
          acc[mt2 * 2 + 1] = wmma_bf16(a, b1, acc[mt2 * 2 + 1]);
        }
      }
#pragma unroll
      for (int mt2 = 0; mt2 < 2; ++mt2)
#pragma unroll
        for (int u = 0; u < 2; ++u)
#pragma unroll
          for (int v = 0; v < 8; ++v)
            gates[(16 * (mg * 2 + mt2) + v + 8 * h) * 1024 + ncol0 + u * 16 + m] =
                acc[mt2 * 2 + u][v];
    }
    __syncthreads();

    // ---- LN over 1024 + add x_gates ----
    {
      float su = 0.f, ssq = 0.f;
      const float* rr = gates + b * 1024 + l16 * 64;
      for (int cc = 0; cc < 64; ++cc) { float v = rr[cc]; su += v; ssq += v * v; }
      su = red16(su); ssq = red16(ssq);
      float mean = su * (1.f / 1024.f);
      float rstd = rsqrtf(ssq * (1.f / 1024.f) - mean * mean + 1e-5f);
      const __bf16* xr = xg + (long)(s * 64 + b) * 1024;
      for (int cc = 0; cc < 64; ++cc) {
        int col = l16 * 64 + cc;
        gates[b * 1024 + col] =
            (gates[b * 1024 + col] - mean) * rstd * g_hh[col] + b_hh[col] + (float)xr[col];
      }
    }
    __syncthreads();

    // ---- cell update + LN(c) + h out ----
    float oarr[16];
    float su = 0.f, ssq = 0.f;
#pragma unroll
    for (int i = 0; i < 16; ++i) {
      int j = j0 + i;
      float iv = sigm(gates[b * 1024 + j]);
      float fv = sigm(gates[b * 1024 + 256 + j]);
      float gv = tanhf(gates[b * 1024 + 512 + j]);
      oarr[i]  = sigm(gates[b * 1024 + 768 + j]);
      c[i] = fv * c[i] + iv * gv;
      su += c[i]; ssq += c[i] * c[i];
    }
    su = red16(su); ssq = red16(ssq);
    float mean = su * (1.f / 256.f);
    float rstd = rsqrtf(ssq * (1.f / 256.f) - mean * mean + 1e-5f);
#pragma unroll
    for (int i = 0; i < 16; ++i) {
      int j = j0 + i;
      float cn = (c[i] - mean) * rstd * g_c[j] + b_c[j];
      __bf16 hb = (__bf16)(oarr[i] * tanhf(cn));
      hbuf[b * 256 + j] = hb;
      hs[(long)b * 65536 + s * 256 + j] = hb;
    }
    __syncthreads();
  }
}

// Output GEMM partials: M=64, N=256, K=65536, K split into 16 chunks of 4096.
// One wave per (tile, kchunk); explicit 2-stage software pipeline.
__global__ void __launch_bounds__(256, 1)
outgemm_kernel(const __bf16* __restrict__ hs,
               const __bf16* __restrict__ Wout,
               float* __restrict__ partials) {
  const int tid = threadIdx.x;
  const int w = tid >> 5, lane = tid & 31;
  const int m = lane & 15, h = lane >> 4;
  const int task = blockIdx.x * 8 + w;
  const int tile = task >> 4, kc = task & 15;
  const int mt = tile >> 4, nt = tile & 15;
  const long kbase = (long)kc * 4096;
  v8f acc = {};
  const __bf16* arow = hs + (long)(16 * mt + m) * 65536 + kbase;
  const __bf16* brow = Wout + (long)(16 * nt + m) * 65536 + kbase;
  v16bf a = load_a16(arow + 16 * h);
  v16bf bm = load_b16(brow + 16 * h);
#pragma unroll 4
  for (int ks = 0; ks < 128; ++ks) {
    v16bf an = a, bn = bm;
    if (ks < 127) {
      an = load_a16(arow + (ks + 1) * 32 + 16 * h);
      bn = load_b16(brow + (ks + 1) * 32 + 16 * h);
    }
    __builtin_prefetch(brow + ks * 32 + 2048, 0, 0);   // global_prefetch_b8
    acc = wmma_bf16(a, bm, acc);
    a = an; bm = bn;
  }
  float* pt = partials + (tile * 16 + kc) * 256;
#pragma unroll
  for (int v = 0; v < 8; ++v) pt[(v + 8 * h) * 16 + m] = acc[v];
}

__global__ void __launch_bounds__(256)
outreduce_kernel(const float* __restrict__ partials,
                 const float* __restrict__ b_out,
                 float* __restrict__ out) {
  int b = blockIdx.x, o = threadIdx.x;
  int tile = (b >> 4) * 16 + (o >> 4);
  int m = b & 15, n = o & 15;
  float s = b_out[o];
#pragma unroll
  for (int kc = 0; kc < 16; ++kc)
    s += partials[(tile * 16 + kc) * 256 + m * 16 + n];
  out[b * 256 + o] = sigm(s);
}

// ---------- launch ----------

extern "C" void kernel_launch(void* const* d_in, const int* in_sizes, int n_in,
                              void* d_out, int out_size, void* d_ws, size_t ws_size,
                              hipStream_t stream) {
  const int*   indices = (const int*)d_in[0];
  const float* table   = (const float*)d_in[1];
  const float* W_ih    = (const float*)d_in[2];
  const float* W_hh    = (const float*)d_in[3];
  const float* g_ih    = (const float*)d_in[4];
  const float* b_ih    = (const float*)d_in[5];
  const float* g_hh    = (const float*)d_in[6];
  const float* b_hh    = (const float*)d_in[7];
  const float* g_c     = (const float*)d_in[8];
  const float* b_c     = (const float*)d_in[9];
  const float* W_out   = (const float*)d_in[10];
  const float* b_out   = (const float*)d_in[11];
  const float* h0      = (const float*)d_in[12];
  const float* c0      = (const float*)d_in[13];

  float* out = (float*)d_out;        // [64,256]
  float* emb_out = out + 64 * 256;   // [64,256,256] (tuple 2nd output)

  char* ws = (char*)d_ws;            // ~82 MB used
  __bf16* Wih_bf   = (__bf16*)(ws + 0);          //  0.5 MB
  __bf16* Whh_bf   = (__bf16*)(ws + 524288);     //  0.5 MB
  __bf16* Wout_bf  = (__bf16*)(ws + 1048576);    // 32   MB
  __bf16* emb_sb   = (__bf16*)(ws + 34603008);   //  8   MB  [s*64+b][256]
  __bf16* xg_bf    = (__bf16*)(ws + 42991616);   // 32   MB  [s*64+b][1024]
  __bf16* hs_bf    = (__bf16*)(ws + 76546048);   //  8   MB  [b][s*256+j]
  float*  partials = (float*) (ws + 84934656);   //  1   MB  [tile][kc][16][16]

  cvt_bf16_kernel<<<256, 256, 0, stream>>>(W_ih, Wih_bf, 1024L * 256);
  cvt_bf16_kernel<<<256, 256, 0, stream>>>(W_hh, Whh_bf, 1024L * 256);
  cvt_bf16_kernel<<<2048, 256, 0, stream>>>(W_out, Wout_bf, 256L * 65536);
  gather_kernel<<<64 * 512, 128, 0, stream>>>(indices, table, emb_out, emb_sb);
  xgates_kernel<<<1024, 256, 73728, stream>>>(emb_sb, Wih_bf, g_ih, b_ih, xg_bf);
  lstm_kernel<<<1, 1024, 294912, stream>>>(Whh_bf, xg_bf, g_hh, b_hh,
                                           g_c, b_c, h0, c0, hs_bf);
  outgemm_kernel<<<128, 256, 0, stream>>>(hs_bf, Wout_bf, partials);
  outreduce_kernel<<<64, 256, 0, stream>>>(partials, b_out, out);
}
